// CausalSelfAttention_29188597743866
// MI455X (gfx1250) — matrix operand order repaired and checked by reading, not compiled, not run on hardware
//
#include <hip/hip_runtime.h>
#include <hip/hip_bf16.h>

typedef __attribute__((ext_vector_type(16))) __bf16        v16bf;
typedef __attribute__((ext_vector_type(8)))  float          v8f;
typedef __attribute__((ext_vector_type(16))) unsigned short v16u;
typedef __attribute__((ext_vector_type(8)))  unsigned short v8u;

#define TT    2048
#define CC    1024
#define NHEAD 16
#define HS    64
#define BB    2
#define N2C   2048
#define PSTR  2056  // LDS P row stride (ushorts), 16B-aligned rows + bank spread

__device__ __forceinline__ unsigned short f2bf(float f) {
  unsigned int u = __float_as_uint(f);
  u += 0x7FFFu + ((u >> 16) & 1u);           // round-to-nearest-even
  return (unsigned short)(u >> 16);
}
__device__ __forceinline__ float bf2f(unsigned short u) {
  return __uint_as_float(((unsigned int)u) << 16);
}
__device__ __forceinline__ v16bf combine(v8u lo, v8u hi) {
  v16u t;
#pragma unroll
  for (int i = 0; i < 8; ++i) { t[i] = lo[i]; t[i + 8] = hi[i]; }
  return __builtin_bit_cast(v16bf, t);
}

// ---------------- convert kernels ----------------
__global__ void cvt_x(const float* __restrict__ x,
                      unsigned short* __restrict__ xb,
                      unsigned short* __restrict__ xT) {
  long i = (long)blockIdx.x * blockDim.x + threadIdx.x;   // over B*T*C
  if (i >= (long)BB * TT * CC) return;
  int c = (int)(i % CC);
  long bt = i / CC;
  int t = (int)(bt % TT);
  int b = (int)(bt / TT);
  unsigned short v = f2bf(x[i]);
  xb[i] = v;
  xT[((long)(b * CC + c)) * TT + t] = v;
}

__global__ void cvt_w(const float* __restrict__ W,
                      unsigned short* __restrict__ WbT) {
  long i = (long)blockIdx.x * blockDim.x + threadIdx.x;   // over C*2C, W[k][n]
  if (i >= (long)CC * N2C) return;
  int n = (int)(i % N2C);
  int k = (int)(i / N2C);
  WbT[(long)n * CC + k] = f2bf(W[i]);
}

// ---------------- qk projection GEMM: [4096 x 2048] = xb @ W + b ------------
__global__ __launch_bounds__(256) void qk_gemm(const unsigned short* __restrict__ xb,
                                               const unsigned short* __restrict__ WbT,
                                               const float* __restrict__ bias,
                                               unsigned short* __restrict__ qs,
                                               unsigned short* __restrict__ kbuf) {
  int wid  = blockIdx.x * 8 + (threadIdx.x >> 5);
  int lane = threadIdx.x & 31, l15 = lane & 15, half = lane >> 4;
  int mt = wid >> 7;        // 256 M-tiles
  int nt = wid & 127;       // 128 N-tiles
  const unsigned short* arow = xb  + (long)(mt * 16 + l15) * CC;
  int ncol = nt * 16 + l15;
  const unsigned short* brow = WbT + (long)ncol * CC;
  int aoff = half * 8, boff = half * 8;
  v8f acc = {};
#pragma unroll 4
  for (int k = 0; k < CC; k += 32) {
    v8u alo = *(const v8u*)(arow + k + aoff);
    v8u ahi = *(const v8u*)(arow + k + aoff + 16);
    v8u blo = *(const v8u*)(brow + k + boff);
    v8u bhi = *(const v8u*)(brow + k + boff + 16);
    acc = __builtin_amdgcn_wmma_f32_16x16x32_bf16(false, combine(alo, ahi),
                                                  false, combine(blo, bhi),
                                                  (short)0, acc, false, false);
  }
  float bn = bias[ncol];
#pragma unroll
  for (int r = 0; r < 8; ++r) {
    int grow = mt * 16 + r + half * 8;     // D layout: m = r + 8*half, n = lane&15
    int t = grow & (TT - 1);
    int b = grow >> 11;
    float val = acc[r] + bn;
    if (ncol < CC) {                       // q, pre-scaled by 1/sqrt(64)
      int h = ncol >> 6, d = ncol & 63;
      qs[(((long)(b * NHEAD + h)) * TT + t) * HS + d] = f2bf(val * 0.125f);
    } else {                               // k
      int kc = ncol - CC, h = kc >> 6, d = kc & 63;
      kbuf[(((long)(b * NHEAD + h)) * TT + t) * HS + d] = f2bf(val);
    }
  }
}

// ---------------- fused attention: scores -> softmax -> P @ x ----------------
__global__ __launch_bounds__(256) void attn(const unsigned short* __restrict__ qs,
                                            const unsigned short* __restrict__ kbuf,
                                            const unsigned short* __restrict__ xT,
                                            const float* __restrict__ hw,
                                            float* __restrict__ out) {
  extern __shared__ unsigned char smem_raw[];
  unsigned short* P   = (unsigned short*)smem_raw;                        // [16][PSTR]
  float* stats = (float*)(smem_raw + (size_t)16 * PSTR * sizeof(unsigned short)); // [8][16]
  float* red   = stats + 8 * 16;                                          // [16]

  int lane = threadIdx.x & 31, wave = threadIdx.x >> 5;
  int l15 = lane & 15, half = lane >> 4;
  int b  = blockIdx.x >> 7;
  int t0 = (blockIdx.x & 127) << 4;
  int smax = t0 + 16;                       // causal bound (exclusive)
  int cb0 = wave * 128;                     // this wave's channel range

  float yacc[8][8];
#pragma unroll
  for (int i = 0; i < 8; ++i)
#pragma unroll
    for (int r = 0; r < 8; ++r) yacc[i][r] = 0.f;

  for (int h = 0; h < NHEAD; ++h) {
    const unsigned short* qrow = qs   + (((long)(b * NHEAD + h)) * TT + t0) * HS;
    const unsigned short* kro  = kbuf + ((long)(b * NHEAD + h)) * TT * HS;

    // ---- phase 1: scores for this wave's 256 columns, raw bf16 into LDS ----
    float mloc[8];
#pragma unroll
    for (int r = 0; r < 8; ++r) mloc[r] = -3.0e38f;
#pragma unroll 1
    for (int i = 0; i < 16; ++i) {
      int sb = (wave * 16 + i) * 16;
      v8f acc = {};
      if (sb < smax) {
#pragma unroll
        for (int kk = 0; kk < 2; ++kk) {
          int dbase = kk * 32;
          v8u alo = *(const v8u*)(qrow + l15 * HS + dbase + half * 8);
          v8u ahi = *(const v8u*)(qrow + l15 * HS + dbase + half * 8 + 16);
          const unsigned short* bp = kro + (long)(sb + l15) * HS + dbase + half * 8;
          v8u blo = *(const v8u*)bp;
          v8u bhi = *(const v8u*)(bp + 16);
          acc = __builtin_amdgcn_wmma_f32_16x16x32_bf16(false, combine(alo, ahi),
                                                        false, combine(blo, bhi),
                                                        (short)0, acc, false, false);
        }
        int s = sb + l15;
#pragma unroll
        for (int r = 0; r < 8; ++r)
          if (s > t0 + r + half * 8) acc[r] = -3.0e38f;   // causal mask
      } else {
#pragma unroll
        for (int r = 0; r < 8; ++r) acc[r] = -3.0e38f;
      }
#pragma unroll
      for (int r = 0; r < 8; ++r) {
        mloc[r] = fmaxf(mloc[r], acc[r]);
        P[(r + half * 8) * PSTR + sb + l15] = f2bf(acc[r]);
      }
    }
    // ---- row max: shuffle within 16-lane halves, then across waves in LDS ----
#pragma unroll
    for (int r = 0; r < 8; ++r) {
      float v = mloc[r];
      v = fmaxf(v, __shfl_xor(v, 1, 32));
      v = fmaxf(v, __shfl_xor(v, 2, 32));
      v = fmaxf(v, __shfl_xor(v, 4, 32));
      v = fmaxf(v, __shfl_xor(v, 8, 32));
      if (l15 == 0) stats[wave * 16 + r + half * 8] = v;
    }
    __syncthreads();
    if (threadIdx.x < 16) {
      float v = -3.0e38f;
      for (int w = 0; w < 8; ++w) v = fmaxf(v, stats[w * 16 + threadIdx.x]);
      red[threadIdx.x] = v;
    }
    __syncthreads();
    float rm[8];
#pragma unroll
    for (int r = 0; r < 8; ++r) rm[r] = red[r + half * 8];

    // ---- exp + partial row sums, rewrite P with probabilities (bf16) ----
    float sums[8];
#pragma unroll
    for (int r = 0; r < 8; ++r) sums[r] = 0.f;
#pragma unroll 1
    for (int i = 0; i < 16; ++i) {
      int sb = (wave * 16 + i) * 16;
#pragma unroll
      for (int r = 0; r < 8; ++r) {
        int idx = (r + half * 8) * PSTR + sb + l15;
        float p = exp2f((bf2f(P[idx]) - rm[r]) * 1.4426950408889634f);
        sums[r] += p;
        P[idx] = f2bf(p);
      }
    }
#pragma unroll
    for (int r = 0; r < 8; ++r) {
      float v = sums[r];
      v += __shfl_xor(v, 1, 32);
      v += __shfl_xor(v, 2, 32);
      v += __shfl_xor(v, 4, 32);
      v += __shfl_xor(v, 8, 32);
      if (l15 == 0) stats[wave * 16 + r + half * 8] = v;
    }
    __syncthreads();
    if (threadIdx.x < 16) {
      float v = 0.f;
      for (int w = 0; w < 8; ++w) v += stats[w * 16 + threadIdx.x];
      red[threadIdx.x] = 1.0f / v;          // every row has >=1 valid column
    }
    __syncthreads();
    float inv[8];
#pragma unroll
    for (int r = 0; r < 8; ++r) inv[r] = red[r + half * 8];

    // ---- phase 2: Z = P @ x over causally-needed K, per-wave 128 channels ----
    v8f z[8];
#pragma unroll
    for (int ct = 0; ct < 8; ++ct) z[ct] = (v8f){};
    int ksteps = (smax + 31) >> 5;
    const unsigned short* xbase = xT + (long)b * CC * TT;
#pragma unroll 1
    for (int ks = 0; ks < ksteps; ++ks) {
      int k0 = ks * 32;
      v8u alo = *(const v8u*)(P + l15 * PSTR + k0 + half * 8);
      v8u ahi = *(const v8u*)(P + l15 * PSTR + k0 + half * 8 + 16);
      v16bf a = combine(alo, ahi);
#pragma unroll
      for (int ct = 0; ct < 8; ++ct) {
        const unsigned short* bp = xbase + (long)(cb0 + ct * 16 + l15) * TT + k0 + half * 8;
        v8u blo = *(const v8u*)bp;
        v8u bhi = *(const v8u*)(bp + 16);
        z[ct] = __builtin_amdgcn_wmma_f32_16x16x32_bf16(false, a,
                                                        false, combine(blo, bhi),
                                                        (short)0, z[ct], false, false);
      }
    }
    // ---- fold softmax normalization + per-head channel weight ----
#pragma unroll
    for (int ct = 0; ct < 8; ++ct) {
      float w = hw[h * CC + cb0 + ct * 16 + l15];
#pragma unroll
      for (int r = 0; r < 8; ++r) yacc[ct][r] += w * inv[r] * z[ct][r];
    }
    __syncthreads();
  }

  // ---- store y [B,T,C] f32 ----
#pragma unroll
  for (int ct = 0; ct < 8; ++ct) {
    int cc2 = cb0 + ct * 16 + l15;
#pragma unroll
    for (int r = 0; r < 8; ++r) {
      int t = t0 + r + half * 8;
      out[((long)(b * TT + t)) * CC + cc2] = yacc[ct][r];
    }
  }
}

// ---------------- launcher ----------------
extern "C" void kernel_launch(void* const* d_in, const int* in_sizes, int n_in,
                              void* d_out, int out_size, void* d_ws, size_t ws_size,
                              hipStream_t stream) {
  const float* x    = (const float*)d_in[0];   // [2,2048,1024]
  const float* W    = (const float*)d_in[1];   // [1024,2048]
  const float* bias = (const float*)d_in[2];   // [2048]
  const float* hw   = (const float*)d_in[3];   // [16,1024]
  float* out = (float*)d_out;

  unsigned char* ws = (unsigned char*)d_ws;
  unsigned short* xb   = (unsigned short*)(ws + 0);                 // 8 MB
  unsigned short* xT   = (unsigned short*)(ws + (8ull  << 20));     // 8 MB
  unsigned short* WbT  = (unsigned short*)(ws + (16ull << 20));     // 4 MB
  unsigned short* qs   = (unsigned short*)(ws + (20ull << 20));     // 8 MB
  unsigned short* kbuf = (unsigned short*)(ws + (28ull << 20));     // 8 MB

  cvt_x<<<(BB * TT * CC + 255) / 256, 256, 0, stream>>>(x, xb, xT);
  cvt_w<<<(CC * N2C + 255) / 256, 256, 0, stream>>>(W, WbT);
  qk_gemm<<<4096, 256, 0, stream>>>(xb, WbT, bias, qs, kbuf);

  size_t smem = (size_t)16 * PSTR * sizeof(unsigned short) + (8 * 16 + 16) * sizeof(float);
  attn<<<BB * (TT / 16), 256, smem, stream>>>(qs, kbuf, xT, hw, out);
}
